// ParticleGNNBiggerWithResidual_25812753449155
// MI455X (gfx1250) — matrix operand order repaired and checked by reading, not compiled, not run on hardware
//
#include <hip/hip_runtime.h>
#include <math.h>

// ---------------------------------------------------------------------------
// ParticleGNN (6x GAT blocks + MLP head) for MI455X / gfx1250.
// Dense GEMMs use v_wmma_f32_16x16x32_bf16 (bf16 A/B, f32 accum).
// Sparse attention/aggregation uses f32 global atomics (L2-resident).
//
// Input order (setup_inputs dict, params flattened in jax-pytree sorted-key
// order):
//   0:x[N,16] 1:edge_index[2,E](int) 2:edge_attr[E,1]
//   3..14  : bn{1..6}.{b,g}
//   15..50 : conv{1..6}.{W, W_e, a_dst, a_e, a_src, b}
//   51..56 : mlp0.{W,b} mlp1.{W,b} mlp2.{W,b}
//   57..60 : mlpbn0.{b,g} mlpbn1.{b,g}
//   61..72 : proj{1..6}.{W,b}
// ---------------------------------------------------------------------------

typedef __attribute__((ext_vector_type(16))) __bf16 v16bf;
typedef __attribute__((ext_vector_type(8)))  __bf16 v8bf;
typedef __attribute__((ext_vector_type(8)))  float  v8f;

#define NEG_SLOPE 0.2f
#define BN_EPS 1e-5f

__device__ __forceinline__ void atomicMaxF(float* addr, float val) {
  if (__float_as_int(val) >= 0)
    atomicMax((int*)addr, __float_as_int(val));
  else
    atomicMin((unsigned int*)addr, __float_as_uint(val));
}

// -------------------- bf16 WMMA GEMM: C[N,M] = A[N,Kp] @ Bt[M,Kp]^T ---------
// One wave per 16x16 C tile, 8 waves per block. Kp % 32 == 0, N % 16 == 0,
// M % 16 == 0. Optional epilogue: +bias[col], BN (g,b), ReLU, bf16 mirror out.
__global__ __launch_bounds__(256)
void gemm_bf16_wmma(const __bf16* __restrict__ A, const __bf16* __restrict__ Bt,
                    float* __restrict__ C, __bf16* __restrict__ Cbf,
                    int Nrows, int M, int Kp,
                    const float* __restrict__ bias,
                    const float* __restrict__ bn_g, const float* __restrict__ bn_b,
                    int do_relu)
{
  const int lane = threadIdx.x & 31;
  const int wave = threadIdx.x >> 5;
  const int mt = M >> 4;
  const int tiles = (Nrows >> 4) * mt;
  const int t = blockIdx.x * 8 + wave;
  if (t >= tiles) return;                    // wave-uniform: EXEC stays all-1s
  const int tr = t / mt, tc = t % mt;
  const int half = lane >> 4;                // 0: lanes 0-15, 1: lanes 16-31
  const int l16  = lane & 15;

  const __bf16* arow = A  + (size_t)(tr * 16 + l16) * Kp;  // A row (node)
  const __bf16* brow = Bt + (size_t)(tc * 16 + l16) * Kp;  // W^T row (out col)

  v8f acc = {};
  for (int kb = 0; kb < Kp; kb += 32) {
    union { v16bf v; struct { v8bf lo; v8bf hi; } p; } a, b;
    // A 16x32 fragment: 16 contiguous K values of row l16
    a.v    = *(const v16bf*)(arow + kb + half * 16);
    // B 32x16 fragment: 16 contiguous K values of column l16
    b.v    = *(const v16bf*)(brow + kb + half * 16);
    acc = __builtin_amdgcn_wmma_f32_16x16x32_bf16(
        false, a.v, false, b.v, (short)0, acc, false, false);
  }

  const int col = tc * 16 + l16;
  const float add = bias ? bias[col] : 0.f;
  float s = 1.f, sh = 0.f;
  if (bn_g) { s = bn_g[col] * rsqrtf(1.f + BN_EPS); sh = bn_b[col]; }
  const int row0 = tr * 16 + half * 8;
#pragma unroll
  for (int g = 0; g < 8; ++g) {
    float v = acc[g] + add;
    if (bn_g) v = v * s + sh;
    if (do_relu) v = v > 0.f ? v : 0.f;
    const size_t idx = (size_t)(row0 + g) * M + col;
    if (C)   C[idx]   = v;
    if (Cbf) Cbf[idx] = (__bf16)v;
  }
}

// -------------------- prep / elementwise kernels ---------------------------
__global__ __launch_bounds__(256)
void cast_pad_rows(const float* __restrict__ X, __bf16* __restrict__ Xb,
                   int Nrows, int K, int Kp) {
  int i = blockIdx.x * 256 + threadIdx.x;
  if (i >= Nrows * Kp) return;
  int r = i / Kp, c = i % Kp;
  Xb[i] = (__bf16)(c < K ? X[(size_t)r * K + c] : 0.f);
}

__global__ __launch_bounds__(256)
void prep_wt(const float* __restrict__ W, __bf16* __restrict__ Wt,
             int K, int M, int Kp) {   // Wt[m*Kp+k] = W[k*M+m], zero-padded
  int i = blockIdx.x * 256 + threadIdx.x;
  if (i >= M * Kp) return;
  int m = i / Kp, k = i % Kp;
  Wt[i] = (__bf16)(k < K ? W[(size_t)k * M + m] : 0.f);
}

__global__ void wedot_kernel(const float* __restrict__ We,
                             const float* __restrict__ ae,
                             float* __restrict__ out, int H, int C) {
  int h = threadIdx.x;
  if (h >= H) return;
  float s = 0.f;
  for (int c = 0; c < C; ++c) s += We[h * C + c] * ae[h * C + c];
  out[h] = s;
}

__global__ __launch_bounds__(256)
void init_nh(float* __restrict__ amax, float* __restrict__ den, int cnt) {
  int i = blockIdx.x * 256 + threadIdx.x;
  if (i < cnt) { amax[i] = -INFINITY; den[i] = 0.f; }
}

__global__ __launch_bounds__(256)
void node_attn(const float* __restrict__ Hb, const float* __restrict__ asrcw,
               const float* __restrict__ adstw, float* __restrict__ asrc,
               float* __restrict__ adst, int Nn, int H, int C) {
  int i = blockIdx.x * 256 + threadIdx.x;
  if (i >= Nn * H) return;
  int n = i / H, h = i % H;
  const float* row = Hb + (size_t)n * (H * C) + h * C;
  float s1 = 0.f, s2 = 0.f;
  for (int c = 0; c < C; ++c) {
    float v = row[c];
    s1 += v * asrcw[h * C + c];
    s2 += v * adstw[h * C + c];
  }
  asrc[i] = s1; adst[i] = s2;
}

__global__ __launch_bounds__(256)
void deg_kernel(const int* __restrict__ ei, const float* __restrict__ ea,
                float* __restrict__ deg, float* __restrict__ easum, int E) {
  int e = blockIdx.x * 256 + threadIdx.x;
  if (e >= E) return;
  int d = ei[E + e];
  atomicAdd(&deg[d], 1.f);
  atomicAdd(&easum[d], ea[e]);
}

__global__ __launch_bounds__(256)
void eaself_kernel(const float* __restrict__ deg, const float* __restrict__ easum,
                   float* __restrict__ eaself, int Nn) {
  int n = blockIdx.x * 256 + threadIdx.x;
  if (n < Nn) eaself[n] = easum[n] / fmaxf(deg[n], 1.f);
}

__global__ __launch_bounds__(256)
void edge_alpha(const int* __restrict__ ei, const float* __restrict__ ea,
                const float* __restrict__ eaself, const float* __restrict__ asrc,
                const float* __restrict__ adst, const float* __restrict__ wedot,
                float* __restrict__ alpha, float* __restrict__ amax,
                int E, int Nn, int H) {
  int e = blockIdx.x * 256 + threadIdx.x;
  if (e >= E + Nn) return;
  int s, d; float eav;
  if (e < E) { s = ei[e]; d = ei[E + e]; eav = ea[e]; }
  else       { s = d = e - E;            eav = eaself[e - E]; }
  for (int h = 0; h < H; ++h) {
    float al = asrc[s * H + h] + adst[d * H + h] + eav * wedot[h];
    al = al > 0.f ? al : NEG_SLOPE * al;
    alpha[(size_t)e * H + h] = al;
    atomicMaxF(&amax[d * H + h], al);
  }
}

__global__ __launch_bounds__(256)
void edge_exp(const int* __restrict__ ei, float* __restrict__ alpha,
              const float* __restrict__ amax, float* __restrict__ den,
              int E, int Nn, int H) {
  int e = blockIdx.x * 256 + threadIdx.x;
  if (e >= E + Nn) return;
  int d = (e < E) ? ei[E + e] : (e - E);
  for (int h = 0; h < H; ++h) {
    float w = __expf(alpha[(size_t)e * H + h] - amax[d * H + h]);
    alpha[(size_t)e * H + h] = w;               // reuse buffer as exp weights
    atomicAdd(&den[d * H + h], w);
  }
}

__global__ __launch_bounds__(256)
void msg_agg(const int* __restrict__ ei, const float* __restrict__ Hb,
             const float* __restrict__ wbuf, const float* __restrict__ den,
             float* __restrict__ agg, int E, int Nn, int H, int C) {
  int t = blockIdx.x * 256 + threadIdx.x;   // over (E+Nn)*H
  if (t >= (E + Nn) * H) return;
  int e = t / H, h = t % H;
  int s, d;
  if (e < E) { s = ei[e]; d = ei[E + e]; } else { s = d = e - E; }
  float w = wbuf[t] / (den[d * H + h] + 1e-16f);
  const float* hr = Hb  + (size_t)s * (H * C) + h * C;
  float*       ar = agg + (size_t)d * (H * C) + h * C;
  for (int c = 0; c < C; ++c) atomicAdd(&ar[c], hr[c] * w);
}

// out = relu( bn(agg + b_conv) + hproj ), written directly in bf16 for the
// next layer's WMMA A operand.
__global__ __launch_bounds__(256)
void combine(const float* __restrict__ agg, const float* __restrict__ hproj,
             const float* __restrict__ bconv, const float* __restrict__ g,
             const float* __restrict__ bb, __bf16* __restrict__ xout,
             int Nn, int M) {
  int i = blockIdx.x * 256 + threadIdx.x;
  if (i >= Nn * M) return;
  int k = i % M;
  float v = (agg[i] + bconv[k]) * (g[k] * rsqrtf(1.f + BN_EPS)) + bb[k] + hproj[i];
  xout[i] = (__bf16)(v > 0.f ? v : 0.f);
}

__global__ __launch_bounds__(256)
void mlp2_lsm(const float* __restrict__ X, const float* __restrict__ W,
              const float* __restrict__ b, float* __restrict__ out, int Nn) {
  int n = blockIdx.x * 256 + threadIdx.x;
  if (n >= Nn) return;
  float z[5];
  for (int j = 0; j < 5; ++j) z[j] = b[j];
  const float* xr = X + (size_t)n * 64;
  for (int k = 0; k < 64; ++k) {
    float xv = xr[k];
    for (int j = 0; j < 5; ++j) z[j] += xv * W[k * 5 + j];
  }
  float m = z[0];
  for (int j = 1; j < 5; ++j) m = fmaxf(m, z[j]);
  float ssum = 0.f;
  for (int j = 0; j < 5; ++j) ssum += __expf(z[j] - m);
  float l = logf(ssum);
  for (int j = 0; j < 5; ++j) out[(size_t)n * 5 + j] = z[j] - m - l;
}

// ---------------------------------------------------------------------------
static inline int cdiv(long long a, int b) { return (int)((a + b - 1) / b); }

extern "C" void kernel_launch(void* const* d_in, const int* in_sizes, int n_in,
                              void* d_out, int out_size, void* d_ws, size_t ws_size,
                              hipStream_t stream) {
  (void)n_in; (void)out_size; (void)ws_size;
  const int Nn = in_sizes[0] / 16;       // 50000
  const int E  = in_sizes[2];            // 320000
  const int ET = E + Nn;

  const float* x  = (const float*)d_in[0];
  const int*   ei = (const int*)d_in[1];
  const float* ea = (const float*)d_in[2];
#define P(i) ((const float*)d_in[(i)])

  // ---- carve workspace ----
  char* wptr = (char*)d_ws;
  auto carve = [&](size_t bytes) {
    void* p = wptr; wptr += (bytes + 255) & ~(size_t)255; return p;
  };
  const size_t bigF = (size_t)Nn * 512 * sizeof(float);
  float*  hbuf   = (float*) carve(bigF);
  float*  hproj  = (float*) carve(bigF);
  float*  agg    = (float*) carve(bigF);
  __bf16* xbf    = (__bf16*)carve((size_t)Nn * 512 * 2);
  __bf16* xbf2   = (__bf16*)carve((size_t)Nn * 128 * 2);
  __bf16* wt     = (__bf16*)carve((size_t)512 * 512 * 2);
  float*  asrc   = (float*) carve((size_t)Nn * 4 * 4);
  float*  adst   = (float*) carve((size_t)Nn * 4 * 4);
  float*  amax   = (float*) carve((size_t)Nn * 4 * 4);
  float*  den    = (float*) carve((size_t)Nn * 4 * 4);
  float*  alpha  = (float*) carve((size_t)ET * 4 * 4);
  float*  wedot  = (float*) carve(256);
  float*  deg    = (float*) carve((size_t)Nn * 4);
  float*  easum  = (float*) carve((size_t)Nn * 4);
  float*  eaself = (float*) carve((size_t)Nn * 4);

  // ---- self-loop edge_attr (fill_value='mean'), once ----
  (void)hipMemsetAsync(deg,   0, (size_t)Nn * 4, stream);
  (void)hipMemsetAsync(easum, 0, (size_t)Nn * 4, stream);
  deg_kernel<<<cdiv(E, 256), 256, 0, stream>>>(ei, ea, deg, easum, E);
  eaself_kernel<<<cdiv(Nn, 256), 256, 0, stream>>>(deg, easum, eaself, Nn);

  // ---- initial activation: f32[N,16] -> bf16[N,32] (K padded to 32) ----
  cast_pad_rows<<<cdiv((long long)Nn * 32, 256), 256, 0, stream>>>(x, xbf, Nn, 16, 32);

  struct Layer { int Kin, Kp, H, C; };
  const Layer L[6] = {
    {16, 32, 4, 64}, {256, 256, 4, 64}, {256, 256, 4, 128},
    {512, 512, 4, 128}, {512, 512, 4, 64}, {256, 256, 1, 64},
  };

  for (int li = 0; li < 6; ++li) {
    const int Kin = L[li].Kin, Kp = L[li].Kp, H = L[li].H, C = L[li].C;
    const int M = H * C;
    const int cw = 15 + li * 6;              // conv.{W, W_e, a_dst, a_e, a_src, b}
    const int bnb = 3 + li * 2, bng = 4 + li * 2;
    const int pw = 61 + li * 2, pb = 62 + li * 2;
    const int tiles = (Nn / 16) * (M / 16);
    const int gblocks = cdiv(tiles, 8);

    // conv GEMM: hbuf = x @ W_conv
    prep_wt<<<cdiv((long long)M * Kp, 256), 256, 0, stream>>>(P(cw), wt, Kin, M, Kp);
    gemm_bf16_wmma<<<gblocks, 256, 0, stream>>>(xbf, wt, hbuf, nullptr, Nn, M, Kp,
                                                nullptr, nullptr, nullptr, 0);
    // residual projection GEMM: hproj = x @ W_proj + b_proj
    prep_wt<<<cdiv((long long)M * Kp, 256), 256, 0, stream>>>(P(pw), wt, Kin, M, Kp);
    gemm_bf16_wmma<<<gblocks, 256, 0, stream>>>(xbf, wt, hproj, nullptr, Nn, M, Kp,
                                                P(pb), nullptr, nullptr, 0);

    // attention scalars
    wedot_kernel<<<1, 32, 0, stream>>>(P(cw + 1), P(cw + 3), wedot, H, C);
    init_nh<<<cdiv(Nn * H, 256), 256, 0, stream>>>(amax, den, Nn * H);
    (void)hipMemsetAsync(agg, 0, (size_t)Nn * M * sizeof(float), stream);
    node_attn<<<cdiv(Nn * H, 256), 256, 0, stream>>>(hbuf, P(cw + 4), P(cw + 2),
                                                     asrc, adst, Nn, H, C);
    edge_alpha<<<cdiv(ET, 256), 256, 0, stream>>>(ei, ea, eaself, asrc, adst, wedot,
                                                  alpha, amax, E, Nn, H);
    edge_exp<<<cdiv(ET, 256), 256, 0, stream>>>(ei, alpha, amax, den, E, Nn, H);
    msg_agg<<<cdiv((long long)ET * H, 256), 256, 0, stream>>>(ei, hbuf, alpha, den,
                                                              agg, E, Nn, H, C);
    // relu(bn(agg + b) + hproj) -> next activations (bf16, no pad needed)
    combine<<<cdiv((long long)Nn * M, 256), 256, 0, stream>>>(
        agg, hproj, P(cw + 5), P(bng), P(bnb), xbf, Nn, M);
  }

  // ---- MLP head ----
  // mlp0: 64 -> 128, bias + bn + relu, bf16 mirror for next GEMM
  prep_wt<<<cdiv(128 * 64, 256), 256, 0, stream>>>(P(51), wt, 64, 128, 64);
  gemm_bf16_wmma<<<cdiv((Nn / 16) * 8, 8), 256, 0, stream>>>(
      xbf, wt, hbuf, xbf2, Nn, 128, 64, P(52), P(58), P(57), 1);
  // mlp1: 128 -> 64, bias + bn + relu (f32 out)
  prep_wt<<<cdiv(64 * 128, 256), 256, 0, stream>>>(P(53), wt, 128, 64, 128);
  gemm_bf16_wmma<<<cdiv((Nn / 16) * 4, 8), 256, 0, stream>>>(
      xbf2, wt, hbuf, nullptr, Nn, 64, 128, P(54), P(60), P(59), 1);
  // mlp2 (64 -> 5) + log_softmax
  mlp2_lsm<<<cdiv(Nn, 256), 256, 0, stream>>>(hbuf, P(55), P(56), (float*)d_out, Nn);
#undef P
}
